// AdaptiveSoftmax_57629871178517
// MI455X (gfx1250) — matrix operand order repaired and checked by reading, not compiled, not run on hardware
//
#include <hip/hip_runtime.h>
#include <stdint.h>

// ---------------------------------------------------------------------------
// Types (plain ext-vector types so they can live in unions)
// ---------------------------------------------------------------------------
typedef __attribute__((ext_vector_type(16))) __bf16        v16bf;
typedef __attribute__((ext_vector_type(8)))  float         v8f;
typedef __attribute__((ext_vector_type(4)))  unsigned int  uint4v;
typedef __attribute__((ext_vector_type(4)))  float         float4v;
typedef __attribute__((ext_vector_type(4)))  unsigned short ushort4v;

union Frag { v16bf v; uint4v q[2]; };

__device__ __forceinline__ unsigned short f2bf(float f) {
    unsigned u = __float_as_uint(f);
    unsigned r = u + 0x7FFFu + ((u >> 16) & 1u);   // round-to-nearest-even
    return (unsigned short)(r >> 16);
}

// ---------------------------------------------------------------------------
// f32 -> bf16 conversion (vectorized x4; all sizes are multiples of 4)
// ---------------------------------------------------------------------------
__global__ __launch_bounds__(256) void cvt_f32_bf16_kernel(
    const float* __restrict__ src, unsigned short* __restrict__ dst, long n4) {
    long i = (long)blockIdx.x * blockDim.x + threadIdx.x;
    if (i < n4) {
        float4v v = ((const float4v*)src)[i];
        ushort4v o;
        o.x = f2bf(v.x); o.y = f2bf(v.y); o.z = f2bf(v.z); o.w = f2bf(v.w);
        ((ushort4v*)dst)[i] = o;
    }
}

// ---------------------------------------------------------------------------
// WMMA GEMM:  C[M x N] = A[M x K] * W[N x K]^T   (bf16 in, f32 accumulate)
//
// Block = 256 threads = 8 waves (wave32), arranged 2 (M) x 4 (N).
// Wave tile = 32 x 64 (2 A-frags x 4 B-frags -> 8 WMMAs / K=32 step).
// Block tile = 64 x 256;  grid = (ceil(N/256), M/64).
//
// Branch-free hot loop:
//   - out-of-range B fragments are clamped to row N-16 (loads valid,
//     results discarded; stores guarded once in the epilogue),
//   - K double-buffered in steps of 64 so loads of the next K-slice are in
//     flight while the current slice's 8 WMMAs issue.
//
// K is a template parameter (1024 / 512 / 256) -> const trip counts and
// const inter-fragment strides (immediate-offset addressing).
// Requires: M % 64 == 0, N % 16 == 0, K % 64 == 0.
// ---------------------------------------------------------------------------
#define WMMA_BF16(Afrag, Bfrag, Cacc) \
    __builtin_amdgcn_wmma_f32_16x16x32_bf16(false, (Afrag).v, false, (Bfrag).v, \
                                            (short)0, (Cacc), false, false)

template<int BF16OUT, int K>
__global__ __launch_bounds__(256) void wmma_gemm_nt(
    const unsigned short* __restrict__ A,
    const unsigned short* __restrict__ W,
    float* __restrict__ outF,
    unsigned short* __restrict__ outH,
    int N, long outPitch, long colOff) {

    const int lane = threadIdx.x & 31;
    const int wv   = threadIdx.x >> 5;     // 0..7
    const int wm   = wv >> 2;              // 0..1
    const int wn   = wv & 3;               // 0..3
    const int l15  = lane & 15;
    const int hi   = lane >> 4;            // 0 or 1

    const int mBase = blockIdx.y * 64 + wm * 32;
    const int nBase = blockIdx.x * 256 + wn * 64;

    // Fragment validity (wave-uniform) + clamped row bases for loads.
    bool vf[4];
    int  nc[4];
    #pragma unroll
    for (int j = 0; j < 4; ++j) {
        const int nf = nBase + j * 16;
        vf[j] = nf < N;
        nc[j] = vf[j] ? nf : (N - 16);
    }

    // A: lane -> row (mBase|+16)+l15, K-offset hi*8 ; two 16B runs per frag.
    // B: lane -> weight row nc[j]+l15, K-offset hi*8 ; two 16B runs per frag.
    const unsigned short* ap0 = A + (size_t)(mBase + l15) * K + (hi << 3);
    const unsigned short* ap1 = ap0 + (size_t)16 * K;
    const unsigned short* bp[4];
    #pragma unroll
    for (int j = 0; j < 4; ++j)
        bp[j] = W + (size_t)(nc[j] + l15) * K + (hi << 3);

    const v8f vzero = {};
    v8f c[2][4];
    #pragma unroll
    for (int i = 0; i < 2; ++i)
        #pragma unroll
        for (int j = 0; j < 4; ++j) c[i][j] = vzero;

    Frag aX[2], bX[4];    // even 32-slice
    Frag aY[2], bY[4];    // odd  32-slice

    auto ldA = [&](Frag& f, const unsigned short* p, int k) {
        f.q[0] = *(const uint4v*)(p + k);
        f.q[1] = *(const uint4v*)(p + k + 16);
    };
    auto ldB = [&](Frag& f, const unsigned short* p, int k) {
        f.q[0] = *(const uint4v*)(p + k);
        f.q[1] = *(const uint4v*)(p + k + 16);
    };

    ldA(aX[0], ap0, 0);
    ldA(aX[1], ap1, 0);
    #pragma unroll
    for (int j = 0; j < 4; ++j) ldB(bX[j], bp[j], 0);

    #pragma unroll 1
    for (int k0 = 0; k0 < K; k0 += 64) {
        // issue loads for slice k0+32 (buffer Y)
        ldA(aY[0], ap0, k0 + 32);
        ldA(aY[1], ap1, k0 + 32);
        #pragma unroll
        for (int j = 0; j < 4; ++j) ldB(bY[j], bp[j], k0 + 32);

        // compute slice k0 (buffer X)
        #pragma unroll
        for (int j = 0; j < 4; ++j) {
            c[0][j] = WMMA_BF16(aX[0], bX[j], c[0][j]);
            c[1][j] = WMMA_BF16(aX[1], bX[j], c[1][j]);
        }

        // issue loads for slice k0+64 (buffer X); clamp to 0 on last iter
        // (branch-free; the extra loads land in-bounds and go unused)
        const int kn = (k0 + 64 < K) ? (k0 + 64) : 0;
        ldA(aX[0], ap0, kn);
        ldA(aX[1], ap1, kn);
        #pragma unroll
        for (int j = 0; j < 4; ++j) ldB(bX[j], bp[j], kn);

        // compute slice k0+32 (buffer Y)
        #pragma unroll
        for (int j = 0; j < 4; ++j) {
            c[0][j] = WMMA_BF16(aY[0], bY[j], c[0][j]);
            c[1][j] = WMMA_BF16(aY[1], bY[j], c[1][j]);
        }
    }

    // C/D layout: VGPR r -> (lanes 0-15: M=r, N=lane) (lanes 16-31: M=8+r, N=lane-16)
    const int mS0 = mBase + (hi ? 8 : 0);
    #pragma unroll
    for (int i = 0; i < 2; ++i) {
        const int mS = mS0 + i * 16;
        #pragma unroll
        for (int j = 0; j < 4; ++j) {
            if (!vf[j]) continue;   // wave-uniform
            const long col = colOff + (nBase + j * 16) + l15;
            if (BF16OUT) {
                unsigned short* base = outH + (size_t)mS * outPitch + col;
                #pragma unroll
                for (int r = 0; r < 8; ++r) base[(size_t)r * outPitch] = f2bf(c[i][j][r]);
            } else {
                float* base = outF + (size_t)mS * outPitch + col;
                #pragma unroll
                for (int r = 0; r < 8; ++r) base[(size_t)r * outPitch] = c[i][j][r];
            }
        }
    }
}

// ---------------------------------------------------------------------------
// Cluster logits: cluster[row][c] = dot(x[row], W_head[20000+c]), c in {0,1}
// ---------------------------------------------------------------------------
__global__ __launch_bounds__(256) void cluster_logits_kernel(
    const float* __restrict__ x, const float* __restrict__ Wh,
    float* __restrict__ cluster) {
    __shared__ float s0[256], s1[256];
    const int row = blockIdx.x;
    const int t = threadIdx.x;
    const float* xr = x + (size_t)row * 1024;
    const float* w0 = Wh + (size_t)20000 * 1024;
    const float* w1 = w0 + 1024;
    float p0 = 0.f, p1 = 0.f;
    for (int i = t; i < 1024; i += 256) {
        float v = xr[i];
        p0 += v * w0[i];
        p1 += v * w1[i];
    }
    s0[t] = p0; s1[t] = p1;
    __syncthreads();
    for (int o = 128; o > 0; o >>= 1) {
        if (t < o) { s0[t] += s0[t + o]; s1[t] += s1[t + o]; }
        __syncthreads();
    }
    if (t == 0) {
        cluster[row * 2 + 0] = s0[0];
        cluster[row * 2 + 1] = s1[0];
    }
}

// ---------------------------------------------------------------------------
// In-place log-softmax over a tail segment of each output row, plus the
// cluster-logit bias.  One block (512 threads) per row; online LSE.
// ---------------------------------------------------------------------------
__global__ __launch_bounds__(512) void logsoftmax_kernel(
    float* __restrict__ out, const float* __restrict__ cluster, int seg) {
    const int row = blockIdx.x;
    const long offset = seg ? 60000 : 20000;
    const int  len    = seg ? 140000 : 40000;
    float* p = out + (size_t)row * 200000 + offset;
    const int t = threadIdx.x;

    float m = -INFINITY, s = 0.f;
    for (int i = t; i < len; i += 512) {
        float v = p[i];
        if (v > m) { s *= __expf(m - v); m = v; }
        s += __expf(v - m);
    }
    __shared__ float sm[512], ss[512];
    sm[t] = m; ss[t] = s;
    __syncthreads();
    for (int o = 256; o > 0; o >>= 1) {
        if (t < o) {
            float m2 = sm[t + o], s2 = ss[t + o];
            float M = fmaxf(sm[t], m2);
            ss[t] = ss[t] * __expf(sm[t] - M) + s2 * __expf(m2 - M);
            sm[t] = M;
        }
        __syncthreads();
    }
    const float add = cluster[row * 2 + seg] - (sm[0] + __logf(ss[0]));
    for (int i = t; i < len; i += 512) p[i] += add;
}

// ---------------------------------------------------------------------------
// Host-side orchestration
// ---------------------------------------------------------------------------
extern "C" void kernel_launch(void* const* d_in, const int* in_sizes, int n_in,
                              void* d_out, int out_size, void* d_ws, size_t ws_size,
                              hipStream_t stream) {
    (void)in_sizes; (void)n_in; (void)out_size; (void)ws_size;

    const float* x   = (const float*)d_in[0];   // (1024, 1024)
    const float* Wh  = (const float*)d_in[1];   // (20002, 1024)
    const float* Wp1 = (const float*)d_in[2];   // (512, 1024)
    const float* Wt1 = (const float*)d_in[3];   // (40000, 512)
    const float* Wp2 = (const float*)d_in[4];   // (256, 1024)
    const float* Wt2 = (const float*)d_in[5];   // (140000, 256)
    float* out = (float*)d_out;                 // (1024, 200000)

    // Workspace layout (bytes)
    char* ws = (char*)d_ws;
    const size_t SZ_XB  = 1024UL * 1024 * 2;
    const size_t SZ_WHB = 20000UL * 1024 * 2;
    const size_t SZ_WP1 = 512UL * 1024 * 2;
    const size_t SZ_WT1 = 40000UL * 512 * 2;
    const size_t SZ_WP2 = 256UL * 1024 * 2;
    const size_t SZ_WT2 = 140000UL * 256 * 2;
    const size_t SZ_P1  = 1024UL * 512 * 2;
    const size_t SZ_P2  = 1024UL * 256 * 2;

    size_t off = 0;
    unsigned short* xb   = (unsigned short*)(ws + off); off += SZ_XB;
    unsigned short* whb  = (unsigned short*)(ws + off); off += SZ_WHB;
    unsigned short* wp1b = (unsigned short*)(ws + off); off += SZ_WP1;
    unsigned short* wt1b = (unsigned short*)(ws + off); off += SZ_WT1;
    unsigned short* wp2b = (unsigned short*)(ws + off); off += SZ_WP2;
    unsigned short* wt2b = (unsigned short*)(ws + off); off += SZ_WT2;
    unsigned short* p1b  = (unsigned short*)(ws + off); off += SZ_P1;
    unsigned short* p2b  = (unsigned short*)(ws + off); off += SZ_P2;
    float* cluster       = (float*)(ws + off);

    auto cvt = [&](const float* s, unsigned short* d, long n) {
        long n4 = n / 4;
        int grid = (int)((n4 + 255) / 256);
        cvt_f32_bf16_kernel<<<grid, 256, 0, stream>>>(s, d, n4);
    };

    // 1) f32 -> bf16 staging
    cvt(x,   xb,   1024L * 1024);
    cvt(Wh,  whb,  20000L * 1024);   // only the 20000 head-word rows
    cvt(Wp1, wp1b, 512L * 1024);
    cvt(Wt1, wt1b, 40000L * 512);
    cvt(Wp2, wp2b, 256L * 1024);
    cvt(Wt2, wt2b, 140000L * 256);

    // 2) cluster logits from f32 directly (2 columns)
    cluster_logits_kernel<<<1024, 256, 0, stream>>>(x, Wh, cluster);

    // 3) GEMMs (grid.y = M/64 = 16; grid.x = ceil(N/256))
    // head word logits -> out cols [0, 20000)
    wmma_gemm_nt<0, 1024><<<dim3(79, 16), 256, 0, stream>>>(
        xb, whb, out, nullptr, 20000, 200000L, 0L);
    // proj1 (bf16 out)
    wmma_gemm_nt<1, 1024><<<dim3(2, 16), 256, 0, stream>>>(
        xb, wp1b, nullptr, p1b, 512, 512L, 0L);
    // tail1 -> out cols [20000, 60000)
    wmma_gemm_nt<0, 512><<<dim3(157, 16), 256, 0, stream>>>(
        p1b, wt1b, out, nullptr, 40000, 200000L, 20000L);
    // proj2 (bf16 out)
    wmma_gemm_nt<1, 1024><<<dim3(1, 16), 256, 0, stream>>>(
        xb, wp2b, nullptr, p2b, 256, 256L, 0L);
    // tail2 -> out cols [60000, 200000)
    wmma_gemm_nt<0, 256><<<dim3(547, 16), 256, 0, stream>>>(
        p2b, wt2b, out, nullptr, 140000, 200000L, 60000L);

    // 4) log-softmax (+cluster bias) in place over the two tail segments
    logsoftmax_kernel<<<1024, 512, 0, stream>>>(out, cluster, 0);
    logsoftmax_kernel<<<1024, 512, 0, stream>>>(out, cluster, 1);
}
